// ContrastiveFlowLoss_72541997629820
// MI455X (gfx1250) — matrix operand order repaired and checked by reading, not compiled, not run on hardware
//
#include <hip/hip_runtime.h>
#include <hip/hip_bf16.h>

typedef __attribute__((ext_vector_type(16))) _Float16 v16h;
typedef __attribute__((ext_vector_type(8)))  _Float16 v8h;
typedef __attribute__((ext_vector_type(8)))  float    v8f;

#define DIM     128
#define TILE    16
#define PCOLS   32          // columns per LDS panel
#define PSTRIDE 136         // padded panel row stride (halves): 128 + 8 -> spreads banks
#define BROWS   64          // rows per block
#define SQRT_INV_T 3.7796447f   // sqrt(1/0.07); folded into both operands: WMMA out == sim/T

__global__ void init_kernel(int* cnt, float* total, int* valid) {
    *cnt = 0; *total = 0.0f; *valid = 0;
}

// One block per row: compute ||z||, write f16 normalized+scaled row, set all-ones flag.
__global__ __launch_bounds__(DIM) void prep_kernel(const float* __restrict__ z,
                                                   const float* __restrict__ attr,
                                                   _Float16* __restrict__ zh,
                                                   float* __restrict__ flagF,
                                                   int* __restrict__ cnt,
                                                   int numAttrs) {
    const int row = blockIdx.x;
    const int t   = threadIdx.x;           // 0..127
    const float x = z[(size_t)row * DIM + t];

    __shared__ float red[DIM];
    red[t] = x * x;
    __syncthreads();
    for (int s = DIM / 2; s > 0; s >>= 1) {
        if (t < s) red[t] += red[t + s];
        __syncthreads();
    }
    const float norm = sqrtf(red[0]);
    const float inv  = SQRT_INV_T / fmaxf(norm, 1e-12f);
    zh[(size_t)row * DIM + t] = (_Float16)(x * inv);

    if (t == 0) {
        bool all1 = true;
        for (int k = 0; k < numAttrs; ++k)
            all1 = all1 && (attr[(size_t)row * numAttrs + k] == 1.0f);
        flagF[row] = all1 ? 1.0f : 0.0f;
        if (all1) atomicAdd(cnt, 1);
    }
}

// 64 rows per block, 8 waves = 4 row-subtiles x 2 column-halves.
// Column panels (32 cols x 128 k, f16) are double-buffered in LDS and shared by
// all waves -> 8x less L2 traffic than per-wave global streaming.
// 4x v_wmma_f32_16x16x32_f16 per 16x16 tile (two independent chains).
// Diagonal tile handling is peeled via a scalarized wave-uniform branch.
__global__ __launch_bounds__(256) void simloss_kernel(const _Float16* __restrict__ zh,
                                                      const float* __restrict__ flagF,
                                                      const int* __restrict__ cntPtr,
                                                      float* __restrict__ totalLoss,
                                                      int* __restrict__ validCnt,
                                                      int B) {
    __shared__ _Float16 panel[2][PCOLS * PSTRIDE];   // 2 x 8704 B
    __shared__ float sAll[BROWS];
    __shared__ float sPos[BROWS];

    const int tid   = threadIdx.x;
    const int lane  = tid & 31;
    const int wave  = tid >> 5;           // 0..7
    const int hi    = lane >> 4;
    const int l15   = lane & 15;
    const int wc    = wave & 1;           // column half within panel
    const int wr    = wave >> 1;          // row subtile 0..3
    const int rowBase = blockIdx.x * BROWS;
    const int myRow16 = rowBase + wr * TILE;
    const int NPAN  = B / PCOLS;

    // ---- A fragments (hoisted): row = myRow16 + l15
    // A k order used here (contiguous, same as B): lanes 0-15 hold K 0..15; lanes 16-31 hold K 16..31
    const _Float16* zrow = zh + (size_t)(myRow16 + l15) * DIM;
    v16h afrag[4];
    #pragma unroll
    for (int kk = 0; kk < 4; ++kk) {
        const int ka = kk * 32 + (hi ? 16 : 0);
        v8h lo  = *(const v8h*)(zrow + ka);
        v8h hip = *(const v8h*)(zrow + ka + 8);
        afrag[kk] = __builtin_shufflevector(lo, hip, 0,1,2,3,4,5,6,7,8,9,10,11,12,13,14,15);
    }

    // Row flags + diagonal selector for this lane's 8 rows
    float rf[8], diagSel[8];
    #pragma unroll
    for (int v = 0; v < 8; ++v) {
        rf[v]      = flagF[myRow16 + 8 * hi + v];
        diagSel[v] = (l15 == (v + 8 * hi)) ? 1.0f : 0.0f;
    }

    // Panel index containing this wave's diagonal tile. It is wave-uniform by
    // construction; readfirstlane lets the compiler emit a scalar branch
    // (s_cmp/s_cbranch) instead of EXEC-mask divergence.
    const int gRowTile  = blockIdx.x * 4 + wr;                       // 16-row tile index
    const int diagPanel = __builtin_amdgcn_readfirstlane(
        ((gRowTile & 1) == wc) ? (gRowTile >> 1) : -1);

    float allAcc[8], posAcc[8];
    #pragma unroll
    for (int v = 0; v < 8; ++v) { allAcc[v] = 0.0f; posAcc[v] = 0.0f; }

    // Cooperative panel copy: thread -> (column c2, 16-half segment seg)
    const int c2  = tid >> 3;     // 0..31
    const int seg = tid & 7;      // 0..7

    // Preload + stage panel 0
    v8h pr0, pr1;
    {
        const _Float16* g = zh + (size_t)c2 * DIM + seg * 16;
        pr0 = *(const v8h*)g;
        pr1 = *(const v8h*)(g + 8);
    }
    {
        _Float16* ld = &panel[0][c2 * PSTRIDE + seg * 16];
        *(v8h*)ld = pr0;
        *(v8h*)(ld + 8) = pr1;
    }

    int buf = 0;
    for (int p = 0; p < NPAN; ++p) {
        __syncthreads();                       // panel[buf] ready; panel[buf^1] free
        const bool more = (p + 1 < NPAN);
        if (more) {                            // issue next panel's global loads early
            const _Float16* g = zh + (size_t)((p + 1) * PCOLS + c2) * DIM + seg * 16;
            pr0 = *(const v8h*)g;
            pr1 = *(const v8h*)(g + 8);
        }

        const float cf = flagF[p * PCOLS + wc * TILE + l15];   // column flag (N = l15)

        // B fragments from LDS: col = wc*16 + l15; lanes 0-15 K 0..15, 16-31 K 16..31.
        // Issue all 8 ds_load_b128 first; fence so they are not sunk between WMMAs.
        const _Float16* lcol = &panel[buf][(wc * TILE + l15) * PSTRIDE];
        const int kb = hi ? 16 : 0;
        v16h b0 = *(const v16h*)(lcol + kb);
        v16h b1 = *(const v16h*)(lcol + 32 + kb);
        v16h b2 = *(const v16h*)(lcol + 64 + kb);
        v16h b3 = *(const v16h*)(lcol + 96 + kb);
        __builtin_amdgcn_sched_barrier(0);

        v8f c0 = {}, c1 = {};
        c0 = __builtin_amdgcn_wmma_f32_16x16x32_f16(false, afrag[0], false, b0, (short)0, c0, false, false);
        c1 = __builtin_amdgcn_wmma_f32_16x16x32_f16(false, afrag[1], false, b1, (short)0, c1, false, false);
        c0 = __builtin_amdgcn_wmma_f32_16x16x32_f16(false, afrag[2], false, b2, (short)0, c0, false, false);
        c1 = __builtin_amdgcn_wmma_f32_16x16x32_f16(false, afrag[3], false, b3, (short)0, c1, false, false);

        if (p == diagPanel) {                  // scalar branch: taken once per wave
            #pragma unroll
            for (int v = 0; v < 8; ++v) {
                const float e = __expf(c0[v] + c1[v]);
                const float m = 1.0f - diagSel[v];
                allAcc[v] = fmaf(m, e, allAcc[v]);
                posAcc[v] = fmaf(m * rf[v] * cf, e, posAcc[v]);
            }
        } else {                               // lean epilogue (255/256 iterations)
            #pragma unroll
            for (int v = 0; v < 8; ++v) {
                const float e = __expf(c0[v] + c1[v]);
                allAcc[v] += e;
                posAcc[v] = fmaf(rf[v] * cf, e, posAcc[v]);
            }
        }

        if (more) {                            // stage next panel into the other buffer
            _Float16* ld = &panel[buf ^ 1][c2 * PSTRIDE + seg * 16];
            *(v8h*)ld = pr0;
            *(v8h*)(ld + 8) = pr1;
        }
        buf ^= 1;
    }

    // Reduce into per-row sums via LDS float atomics.
    if (tid < BROWS) { sAll[tid] = 0.0f; sPos[tid] = 0.0f; }
    __syncthreads();
    #pragma unroll
    for (int v = 0; v < 8; ++v) {
        const int r = wr * TILE + v + 8 * hi;
        atomicAdd(&sAll[r], allAcc[v]);
        atomicAdd(&sPos[r], posAcc[v]);
    }
    __syncthreads();

    if (tid < BROWS) {
        const int i   = rowBase + tid;
        const int cnt = *cntPtr;                       // number of all-ones rows
        const float fi = flagF[i];
        const float num_pos = (fi > 0.0f) ? (float)(cnt - 1) : 0.0f;
        const float all_sum = sAll[tid];               // diagonal already excluded
        // masked entries contribute exp(0)=1 each (incl. diagonal): B - num_pos of them
        const float pos_sum = ((float)B - num_pos) + sPos[tid];
        const float loss_i  = logf(all_sum) - logf(fmaxf(pos_sum, 1e-12f));
        const bool valid = (num_pos > 0.0f) && (all_sum > 0.0f) && (pos_sum > 0.0f);
        if (valid) {
            atomicAdd(totalLoss, loss_i);
            atomicAdd(validCnt, 1);
        }
    }
}

__global__ void fin_kernel(const float* total, const int* valid, float* out) {
    out[0] = (*valid > 0) ? (*total / (float)(*valid)) : 0.0f;
}

extern "C" void kernel_launch(void* const* d_in, const int* in_sizes, int n_in,
                              void* d_out, int out_size, void* d_ws, size_t ws_size,
                              hipStream_t stream) {
    const float* z    = (const float*)d_in[0];   // [B, 128] f32
    const float* attr = (const float*)d_in[1];   // [B, numAttrs] f32

    const int B        = in_sizes[0] / DIM;
    const int numAttrs = in_sizes[1] / B;

    char* ws = (char*)d_ws;
    _Float16* zh = (_Float16*)ws;
    size_t off = (size_t)B * DIM * sizeof(_Float16);          // 2 MB @ B=8192
    float* flagF = (float*)(ws + off); off += (size_t)B * sizeof(float);
    off = (off + 15) & ~(size_t)15;
    int*   cnt       = (int*)(ws + off);  off += 16;
    float* totalLoss = (float*)(ws + off); off += 16;
    int*   validCnt  = (int*)(ws + off);

    init_kernel<<<1, 1, 0, stream>>>(cnt, totalLoss, validCnt);
    prep_kernel<<<B, DIM, 0, stream>>>(z, attr, zh, flagF, cnt, numAttrs);
    simloss_kernel<<<B / BROWS, 256, 0, stream>>>(zh, flagF, cnt, totalLoss, validCnt, B);
    fin_kernel<<<1, 1, 0, stream>>>(totalLoss, validCnt, (float*)d_out);
}
